// FallbackGRU_57870389346866
// MI455X (gfx1250) — matrix operand order repaired and checked by reading, not compiled, not run on hardware
//
#include <hip/hip_runtime.h>
#include <math.h>

typedef __attribute__((ext_vector_type(16))) __bf16 v16bf;
typedef __attribute__((ext_vector_type(8)))  float  v8f;

#define B_TOT 512
#define T_LEN 1024
#define I_DIM 64
#define H_DIM 64

// ---- bf16 helpers: let the compiler pick the gfx1250 hardware cvt ----
static __device__ __forceinline__ __bf16 f2bf(float x){
  return static_cast<__bf16>(x);
}
static __device__ __forceinline__ float bf2f(__bf16 b){
  unsigned short s = __builtin_bit_cast(unsigned short, b);
  unsigned u = ((unsigned)s) << 16;
  return __builtin_bit_cast(float, u);
}
static __device__ __forceinline__ v8f splat8(float v){
  v8f r;
#pragma unroll
  for(int i=0;i<8;i++) r[i]=v;
  return r;
}
static __device__ __forceinline__ v8f wmma_bf16(v16bf a, v16bf b, v8f c){
  // 8 args: (neg_a, A, neg_b, B, c_mod, C, reuse_a, reuse_b)
  return __builtin_amdgcn_wmma_f32_16x16x32_bf16(false, a, false, b, (short)0, c, false, false);
}

// B-operand tile (32x16 KxN) from fp32 weight W[192][64]; rows g0..g0+15 are the N columns.
// B layout (wave32, 16-bit): lanes 0-15: elts0-7=K0..7, elts8-15=K16..23; lanes 16-31: K8..15 / K24..31.
static __device__ __forceinline__ v16bf load_Btile(const float* __restrict__ W, int g0, int lane, int khalf){
  int n  = lane & 15;
  int kb = ((lane >> 4) & 1) * 8 + khalf * 32;
  const float* p = W + (g0 + n) * 64 + kb;
  v16bf r;
#pragma unroll
  for(int j=0;j<16;j++) r[j] = f2bf(p[j + (j & 8)]);
  return r;
}

// A-operand tile (16x32 MxK) from LDS bf16 [16][64].
// A layout: lanes 0-15: elts0-7=K0..7, elts8-15=K16..23; lanes 16-31: K8..15 / K24..31.
static __device__ __forceinline__ v16bf load_Atile_lds(const __bf16* hb, int lane, int khalf){
  int m  = lane & 15;
  int kb = ((lane >> 4) & 1) * 8 + khalf * 32;
  const __bf16* p = hb + m * 64 + kb;
  v16bf r;
#pragma unroll
  for(int j=0;j<8;j++){ r[j] = p[j]; r[8+j] = p[16+j]; }
  return r;
}

// A-operand tile from global fp32 x rows (row stride = T*I).
static __device__ __forceinline__ v16bf load_Atile_x(const float* __restrict__ base, long strideRow, int lane, int khalf){
  int m  = lane & 15;
  int kb = ((lane >> 4) & 1) * 8 + khalf * 32;
  const float* p = base + (long)m * strideRow + kb;
  v16bf r;
#pragma unroll
  for(int j=0;j<8;j++){ r[j] = f2bf(p[j]); r[8+j] = f2bf(p[16+j]); }
  return r;
}

// ---- fast gate nonlinearities: v_exp_f32 + v_rcp_f32, no IEEE division chains ----
static __device__ __forceinline__ float fast_sig(float x){
  return __builtin_amdgcn_rcpf(1.f + __expf(-x));
}
static __device__ __forceinline__ float fast_tanh(float x){
  // 1 - 2/(e^{2x}+1); saturates correctly: exp->inf => rcp->0 => +1; exp->0 => -1
  return __builtin_fmaf(-2.f, __builtin_amdgcn_rcpf(__expf(2.f*x) + 1.f), 1.f);
}
static __device__ __forceinline__ v8f vsig(v8f x){
  v8f r;
#pragma unroll
  for(int i=0;i<8;i++) r[i] = fast_sig(x[i]);
  return r;
}
static __device__ __forceinline__ v8f vtanh(v8f x){
  v8f r;
#pragma unroll
  for(int i=0;i<8;i++) r[i] = fast_tanh(x[i]);
  return r;
}

// One workgroup = 4 waves = one 16-row batch tile; wave w owns hidden columns [16w,16w+16)
// and gate tiles {w, 4+w, 8+w} of each [16,192] GEMM. Weights live in registers (bf16 B-layout).
__global__ __launch_bounds__(128) void gru2_wmma_fused(
    const float* __restrict__ x,
    const float* __restrict__ Wih0, const float* __restrict__ Whh0,
    const float* __restrict__ bih0, const float* __restrict__ bhh0,
    const float* __restrict__ Wih1, const float* __restrict__ Whh1,
    const float* __restrict__ bih1, const float* __restrict__ bhh1,
    const float* __restrict__ fcw,  const float* __restrict__ fcb,
    float* __restrict__ out)
{
  __shared__ __bf16 sh0[16*64];   // layer-0 hidden relayout buffer (bf16 [16][64])
  __shared__ __bf16 sh1[16*64];   // layer-1 hidden relayout buffer
  __shared__ float  sfcw[64];

  const int tid  = threadIdx.x;
  const int wave = tid >> 5;
  const int lane = tid & 31;
  const int b0   = blockIdx.x * 16;
  const int c0   = wave * 16;     // this wave's hidden column block

  for(int i = tid; i < 16*64; i += 128){ sh0[i] = f2bf(0.f); sh1[i] = f2bf(0.f); }
  if(tid < 64) sfcw[tid] = fcw[tid];

  // Stage all 4 weight matrices into registers, bf16 B-operand layout.
  v16bf Bg[4][3][2];
  {
    const float* Ws[4] = {Wih0, Whh0, Wih1, Whh1};
#pragma unroll
    for(int mtx=0; mtx<4; mtx++)
#pragma unroll
      for(int g=0; g<3; g++)
#pragma unroll
        for(int kh=0; kh<2; kh++)
          Bg[mtx][g][kh] = load_Btile(Ws[mtx], g*64 + c0, lane, kh);
  }

  // Bias value per accumulator tile: depends only on output column -> one scalar/lane,
  // splat across the 8 C VGPRs (C/D layout has fixed N per lane).
  float bx0[3], bh0g[3], bx1[3], bh1g[3];
  {
    int col = c0 + (lane & 15);
#pragma unroll
    for(int g=0; g<3; g++){
      bx0[g]  = bih0[g*64+col]; bh0g[g] = bhh0[g*64+col];
      bx1[g]  = bih1[g*64+col]; bh1g[g] = bhh1[g*64+col];
    }
  }
  const float fcb0 = fcb[0];

  v8f h0 = splat8(0.f), h1 = splat8(0.f);   // f32 carried state (this wave's columns, D layout)

  const long xstride = (long)T_LEN * I_DIM;
  const float* xb = x + (long)b0 * xstride;

  const int nn    = lane & 15;
  const int mbase = (lane >> 4) * 8;

#pragma unroll 1
  for(int t=0; t<T_LEN; t++){
    // A-operands of previous hidden states (reads must finish before anyone stores: sync below)
    v16bf hA0p0 = load_Atile_lds(sh0, lane, 0);
    v16bf hA0p1 = load_Atile_lds(sh0, lane, 1);
    v16bf hA1p0 = load_Atile_lds(sh1, lane, 0);
    v16bf hA1p1 = load_Atile_lds(sh1, lane, 1);
    __syncthreads();

    const float* xt = xb + (long)t * I_DIM;
    v16bf xA0 = load_Atile_x(xt, xstride, lane, 0);
    v16bf xA1 = load_Atile_x(xt, xstride, lane, 1);
    if(t + 1 < T_LEN){
      // global_prefetch_b8 for next timestep's x row chunk of this lane
      const float* pf = xt + I_DIM + (long)(lane & 15) * xstride;
      __builtin_prefetch(pf, 0, 1);
    }

    // ---- layer 0: gx = x@Wih0^T + bih0 ; gh = h0@Whh0^T + bhh0 (this wave's 3 gate tiles)
    v8f gxa[3], gha[3];
#pragma unroll
    for(int g=0; g<3; g++){
      v8f c = splat8(bx0[g]);
      c = wmma_bf16(xA0, Bg[0][g][0], c);
      c = wmma_bf16(xA1, Bg[0][g][1], c);
      gxa[g] = c;
      v8f d = splat8(bh0g[g]);
      d = wmma_bf16(hA0p0, Bg[1][g][0], d);
      d = wmma_bf16(hA0p1, Bg[1][g][1], d);
      gha[g] = d;
    }
    {
      v8f r = vsig(gxa[0] + gha[0]);
      v8f z = vsig(gxa[1] + gha[1]);
      v8f n = vtanh(gxa[2] + r * gha[2]);
      h0 = n + z * (h0 - n);            // == (1-z)*n + z*h0
    }
    // D-layout -> LDS [16][64] bf16 (this wave writes its 16 columns)
#pragma unroll
    for(int rr=0; rr<8; rr++) sh0[(mbase+rr)*64 + c0 + nn] = f2bf(h0[rr]);
    __syncthreads();

    v16bf hA0n0 = load_Atile_lds(sh0, lane, 0);
    v16bf hA0n1 = load_Atile_lds(sh0, lane, 1);

    // ---- layer 1: gx = h0new@Wih1^T + bih1 ; gh = h1@Whh1^T + bhh1
#pragma unroll
    for(int g=0; g<3; g++){
      v8f c = splat8(bx1[g]);
      c = wmma_bf16(hA0n0, Bg[2][g][0], c);
      c = wmma_bf16(hA0n1, Bg[2][g][1], c);
      gxa[g] = c;
      v8f d = splat8(bh1g[g]);
      d = wmma_bf16(hA1p0, Bg[3][g][0], d);
      d = wmma_bf16(hA1p1, Bg[3][g][1], d);
      gha[g] = d;
    }
    {
      v8f r = vsig(gxa[0] + gha[0]);
      v8f z = vsig(gxa[1] + gha[1]);
      v8f n = vtanh(gxa[2] + r * gha[2]);
      h1 = n + z * (h1 - n);
    }
#pragma unroll
    for(int rr=0; rr<8; rr++) sh1[(mbase+rr)*64 + c0 + nn] = f2bf(h1[rr]);
    __syncthreads();

    // ---- linear head: out[b,t] = h1 . fc_w + fc_b (wave 0; shfl_xor(16) reduction)
    if(wave == 0){
      int m = lane & 15, half = lane >> 4;
      float s = 0.f;
#pragma unroll
      for(int c=0; c<32; c++){
        int cc = half*32 + c;
        s += bf2f(sh1[m*64 + cc]) * sfcw[cc];
      }
      s += __shfl_xor(s, 16, 32);
      if(lane < 16) out[(long)(b0 + m) * T_LEN + t] = s + fcb0;
    }
  }
}

extern "C" void kernel_launch(void* const* d_in, const int* in_sizes, int n_in,
                              void* d_out, int out_size, void* d_ws, size_t ws_size,
                              hipStream_t stream)
{
  (void)in_sizes; (void)n_in; (void)out_size; (void)d_ws; (void)ws_size;
  const float* x    = (const float*)d_in[0];
  const float* Wih0 = (const float*)d_in[1];
  const float* Whh0 = (const float*)d_in[2];
  const float* bih0 = (const float*)d_in[3];
  const float* bhh0 = (const float*)d_in[4];
  const float* Wih1 = (const float*)d_in[5];
  const float* Whh1 = (const float*)d_in[6];
  const float* bih1 = (const float*)d_in[7];
  const float* bhh1 = (const float*)d_in[8];
  const float* fcw  = (const float*)d_in[9];
  const float* fcb  = (const float*)d_in[10];
  float* out = (float*)d_out;

  dim3 grid(B_TOT / 16), block(128);
  hipLaunchKernelGGL(gru2_wmma_fused, grid, block, 0, stream,
                     x, Wih0, Whh0, bih0, bhh0, Wih1, Whh1, bih1, bhh1, fcw, fcb, out);
}
